// Net_44023414784339
// MI455X (gfx1250) — matrix operand order repaired and checked by reading, not compiled, not run on hardware
//
#include <hip/hip_runtime.h>
#include <hip/hip_bf16.h>

typedef __attribute__((ext_vector_type(16))) _Float16 v16h;
typedef __attribute__((ext_vector_type(8)))  float    v8f;

#define CAPS_EPS 1e-4f

__device__ __forceinline__ float sigmoidf(float v){ return 1.0f/(1.0f+expf(-v)); }

// ---------------- utility kernels ----------------
__global__ void zero_f(float* p, long n){
  long i = (long)blockIdx.x*blockDim.x + threadIdx.x;
  long stride = (long)gridDim.x*blockDim.x;
  for (; i < n; i += stride) p[i] = 0.0f;
}

__global__ void init_mm(float* mm){
  unsigned* u = (unsigned*)mm;
  u[0] = 0xFFFFFFFFu;  // min key seed
  u[1] = 0u;           // max key seed
}

__global__ void minmax_k(const float* __restrict__ p, int n, float* mm){
  int i = blockIdx.x*blockDim.x + threadIdx.x;
  if (i >= n) return;
  unsigned b = __float_as_uint(p[i]);
  unsigned key = (b & 0x80000000u) ? ~b : (b | 0x80000000u); // monotone map
  unsigned* u = (unsigned*)mm;
  atomicMin(&u[0], key);
  atomicMax(&u[1], key);
}

__global__ void mm_final(float* mm){
  unsigned* u = (unsigned*)mm;
  unsigned kmin = u[0], kmax = u[1];
  unsigned bmin = (kmin & 0x80000000u) ? (kmin & 0x7FFFFFFFu) : ~kmin;
  unsigned bmax = (kmax & 0x80000000u) ? (kmax & 0x7FFFFFFFu) : ~kmax;
  mm[2] = __uint_as_float(bmin);
  mm[3] = __uint_as_float(bmax);
}

// ---------------- spline conv via WMMA ----------------
// Per wave: stage a 16-edge x 128-kpos f16 basis tile in LDS (8 nonzeros/row),
// multiply against LDS-resident f16 W (125x16 padded to 128) with
// 4x v_wmma_f32_16x16x32_f16, then scatter x[src]*wk into agg via f32 atomics.
__global__ __launch_bounds__(256) void spline_wmma(
    const float* __restrict__ x, const float* __restrict__ pseudo,
    const float* __restrict__ Wsp, const int* __restrict__ eidx,
    float* __restrict__ agg, float* __restrict__ deg, int E)
{
  __shared__ _Float16 Wh[2048];        // Wh[n*128 + k]  (B operand, column-major-ish)
  __shared__ _Float16 Stile[8][2048];  // per-wave basis tile [row*128 + k]

  const int tid = threadIdx.x;
  for (int i = tid; i < 2048; i += 256){
    int n = i >> 7, k = i & 127;
    float w = (k < 125) ? Wsp[k*16 + n] : 0.0f;
    Wh[i] = (_Float16)w;
  }
  __syncthreads();

  const int wave = tid >> 5;
  const int lane = tid & 31;
  const int tile = blockIdx.x*8 + wave;
  const int base = tile*16;
  _Float16* Sw = &Stile[wave][0];

  { // zero this wave's tile (1024 dwords / 32 lanes)
    unsigned* Sz = (unsigned*)Sw;
    #pragma unroll
    for (int i = 0; i < 32; ++i) Sz[lane + i*32] = 0u;
  }
  __syncthreads();  // barrier also orders type-punned LDS writes

  const int* srcp = eidx;
  const int* dstp = eidx + E;

  if (lane < 16){
    int e = base + lane;
    if (e < E){
      __builtin_prefetch(&pseudo[(size_t)e*3 + 6144], 0, 1);
      int   i0[3];
      float fr[3];
      #pragma unroll
      for (int d = 0; d < 3; ++d){
        float vv = pseudo[e*3+d]*4.0f;                  // * (K-1)
        float fl = fminf(fmaxf(floorf(vv), 0.0f), 3.0f);
        i0[d] = (int)fl; fr[d] = vv - fl;
      }
      #pragma unroll
      for (int b = 0; b < 8; ++b){
        int b0=(b>>2)&1, b1=(b>>1)&1, b2=b&1;
        int kf = ((i0[0]+b0)*5 + (i0[1]+b1))*5 + (i0[2]+b2);   // < 125
        float wb = (b0?fr[0]:1.0f-fr[0])*(b1?fr[1]:1.0f-fr[1])*(b2?fr[2]:1.0f-fr[2]);
        Sw[lane*128 + kf] = (_Float16)wb;
      }
      atomicAdd(&deg[dstp[e]], 1.0f);
    }
  }
  __syncthreads();

  const int hi = (lane >> 4) & 1;
  const int rc = lane & 15;          // A row == B column == C column

  v8f acc = {0.f,0.f,0.f,0.f,0.f,0.f,0.f,0.f};
  #pragma unroll
  for (int ch = 0; ch < 4; ++ch){
    const int kbase = ch*32;
    union { unsigned u[8]; v16h h; } Af, Bf;
    #pragma unroll
    for (int i = 0; i < 8; ++i){
      // A 16x32 f16 layout: v0..3 = K{0..7}(+8 for hi lanes), v4..7 = K{16..23}(+8)
      int klo = ((i < 4) ? 2*i : 16 + 2*(i-4)) + hi*8;
      Af.u[i] = *reinterpret_cast<const unsigned*>(&Sw[rc*128 + kbase + klo]);
      // B 32x16 f16 layout: v0..3 = K{0..7}(+8 for hi lanes), v4..7 = K{16..23}(+8), pairs per VGPR
      int kk  = kbase + ((i < 4) ? 2*i : 16 + 2*(i-4)) + hi*8;
      Bf.u[i] = *reinterpret_cast<const unsigned*>(&Wh[rc*128 + kk]);
    }
    acc = __builtin_amdgcn_wmma_f32_16x16x32_f16(false, Af.h, false, Bf.h,
                                                 (short)0, acc, false, false);
  }

  // C layout: lane holds column rc, rows r + 8*hi
  #pragma unroll
  for (int r = 0; r < 8; ++r){
    int m = r + hi*8;
    int e = base + m;
    if (e < E){
      float val = x[srcp[e]] * acc[r];
      atomicAdd(&agg[dstp[e]*16 + rc], val);
    }
  }
}

// ---------------- node finalize: h -> elu -> lin -> normalized quat ----------------
__global__ void node_finalize(const float* __restrict__ x, const float* __restrict__ agg,
    const float* __restrict__ deg, const float* __restrict__ root, const float* __restrict__ bias,
    const float* __restrict__ lin_w, const float* __restrict__ lin_b,
    float* __restrict__ pose_out, float* __restrict__ act_out, int N)
{
  int n = blockIdx.x*blockDim.x + threadIdx.x;
  if (n >= N) return;
  float xv = x[n];
  float dv = fmaxf(deg[n], 1.0f);
  float q[4] = {lin_b[0], lin_b[1], lin_b[2], lin_b[3]};
  for (int o = 0; o < 16; ++o){
    float h = agg[n*16+o]/dv + xv*root[o] + bias[o];
    float e = (h > 0.0f) ? h : (expf(h) - 1.0f);   // elu
    #pragma unroll
    for (int c = 0; c < 4; ++c) q[c] += e*lin_w[o*4+c];
  }
  float inv = 1.0f/(sqrtf(q[0]*q[0]+q[1]*q[1]+q[2]*q[2]+q[3]*q[3]) + CAPS_EPS);
  #pragma unroll
  for (int c = 0; c < 4; ++c) pose_out[n*4+c] = q[c]*inv;
  act_out[n] = xv;
}

// ---------------- capsule routing layer ----------------
__device__ __forceinline__ void pred_calc(const float* __restrict__ qn4,
                                          const float* __restrict__ r4, float* out)
{
  float qx=qn4[0], qy=qn4[1], qz=qn4[2], qw=qn4[3];
  float rx=r4[0],  ry=r4[1],  rz=r4[2],  rw=r4[3];
  float px = qw*rx + qx*rw + qy*rz - qz*ry;
  float py = qw*ry - qx*rz + qy*rw + qz*rx;
  float pz = qw*rz + qx*ry - qy*rx + qz*rw;
  float pw = qw*rw - qx*rx - qy*ry - qz*rz;
  float inv = 1.0f/(sqrtf(px*px+py*py+pz*pz+pw*pw) + CAPS_EPS);
  out[0]=px*inv; out[1]=py*inv; out[2]=pz*inv; out[3]=pw*inv;
}

template<int I, int J>
__global__ void caps_kernel(int Nn,
    const float* __restrict__ pose_in, const float* __restrict__ act_in,
    const float* __restrict__ qw, float* __restrict__ pose_out, float* __restrict__ act_out)
{
  __shared__ float qn[I*J*4];
  for (int t = threadIdx.x; t < I*J; t += blockDim.x){
    float a=qw[t*4+0], b=qw[t*4+1], c=qw[t*4+2], d=qw[t*4+3];
    float inv = 1.0f/(sqrtf(a*a+b*b+c*c+d*d) + CAPS_EPS);
    qn[t*4+0]=a*inv; qn[t*4+1]=b*inv; qn[t*4+2]=c*inv; qn[t*4+3]=d*inv;
  }
  __syncthreads();
  int n = blockIdx.x*blockDim.x + threadIdx.x;
  if (n >= Nn) return;

  float pose[I][4], actv[I];
  for (int i=0;i<I;++i){
    for (int k=0;k<4;++k) pose[i][k] = pose_in[(n*I+i)*4+k];
    actv[i] = act_in[n*I+i];
  }
  float bb[I][J], cc[I][J], v[J][4], tacc[J];
  for (int i=0;i<I;++i) for (int j=0;j<J;++j) bb[i][j] = 0.0f;

  for (int it = 0; it < 3; ++it){
    for (int i=0;i<I;++i){               // c = softmax_j(b) * act
      float mx = bb[i][0];
      for (int j=1;j<J;++j) mx = fmaxf(mx, bb[i][j]);
      float sum = 0.0f;
      for (int j=0;j<J;++j){ cc[i][j] = expf(bb[i][j]-mx); sum += cc[i][j]; }
      float sc = actv[i]/sum;
      for (int j=0;j<J;++j) cc[i][j] *= sc;
    }
    float s[J][4];
    for (int j=0;j<J;++j) for (int k=0;k<4;++k) s[j][k] = 0.0f;
    for (int i=0;i<I;++i)
      for (int j=0;j<J;++j){
        float pr[4]; pred_calc(&qn[(i*J+j)*4], pose[i], pr);
        for (int k=0;k<4;++k) s[j][k] += cc[i][j]*pr[k];
      }
    for (int j=0;j<J;++j){
      float inv = 1.0f/(sqrtf(s[j][0]*s[j][0]+s[j][1]*s[j][1]+s[j][2]*s[j][2]+s[j][3]*s[j][3]) + CAPS_EPS);
      for (int k=0;k<4;++k) v[j][k] = s[j][k]*inv;
    }
    bool last = (it == 2);
    if (last) for (int j=0;j<J;++j) tacc[j] = 0.0f;
    for (int i=0;i<I;++i)
      for (int j=0;j<J;++j){
        float pr[4]; pred_calc(&qn[(i*J+j)*4], pose[i], pr);
        float dp = pr[0]*v[j][0]+pr[1]*v[j][1]+pr[2]*v[j][2]+pr[3]*v[j][3];
        bb[i][j] += dp;
        if (last) tacc[j] += cc[i][j]*dp;
      }
  }
  for (int j=0;j<J;++j){
    for (int k=0;k<4;++k) pose_out[(n*J+j)*4+k] = v[j][k];
    act_out[n*J+j] = sigmoidf(tacc[j]);
  }
}

// ---------------- voxel pooling ----------------
__device__ __forceinline__ void grid_params(const float* mm, int G, int mode,
                                            float& start, float& size){
  float mn = mm[2], mx = mm[3];
  if (mode == 0){ start = mn;        size = (mx + 0.001f - mn)/(float)G; }
  else          { start = mn - 0.5f; size = mx - mn + 1.0f; }
}
__device__ __forceinline__ int vox_id(const float* pp, float start, float size, int G){
  int id = 0;
  #pragma unroll
  for (int d=0; d<3; ++d){
    float g = floorf((pp[d]-start)/size);
    g = fminf(fmaxf(g, 0.0f), (float)(G-1));
    id = id*G + (int)g;
  }
  return id;
}

__global__ void pool_p1(int Nn, int G, int Jc, int mode, const float* __restrict__ mm,
    const float* __restrict__ pose_in, const float* __restrict__ act_in,
    const float* __restrict__ pos_in,
    float* __restrict__ s, float* __restrict__ wsum,
    float* __restrict__ cnt, float* __restrict__ possum)
{
  int n = blockIdx.x*blockDim.x + threadIdx.x;
  if (n >= Nn) return;
  float start, size; grid_params(mm, G, mode, start, size);
  int vid = vox_id(&pos_in[n*3], start, size, G);
  for (int j=0;j<Jc;++j){
    float a = act_in[n*Jc+j];
    for (int k=0;k<4;++k) atomicAdd(&s[(vid*Jc+j)*4+k], a*pose_in[(n*Jc+j)*4+k]);
    atomicAdd(&wsum[vid*Jc+j], a);
  }
  atomicAdd(&cnt[vid], 1.0f);
  for (int d=0;d<3;++d) atomicAdd(&possum[vid*3+d], pos_in[n*3+d]);
}

__global__ void pool_p2(int total, const float* __restrict__ s, float* __restrict__ m){
  int t = blockIdx.x*blockDim.x + threadIdx.x;
  if (t >= total) return;
  const float* sp = &s[t*4];
  float inv = 1.0f/(sqrtf(sp[0]*sp[0]+sp[1]*sp[1]+sp[2]*sp[2]+sp[3]*sp[3]) + CAPS_EPS);
  for (int k=0;k<4;++k) m[t*4+k] = sp[k]*inv;
}

__global__ void pool_p3(int Nn, int G, int Jc, int mode, const float* __restrict__ mm,
    const float* __restrict__ pose_in, const float* __restrict__ act_in,
    const float* __restrict__ pos_in,
    const float* __restrict__ m, float* __restrict__ asum)
{
  int n = blockIdx.x*blockDim.x + threadIdx.x;
  if (n >= Nn) return;
  float start, size; grid_params(mm, G, mode, start, size);
  int vid = vox_id(&pos_in[n*3], start, size, G);
  for (int j=0;j<Jc;++j){
    float dp = 0.0f;
    for (int k=0;k<4;++k) dp += pose_in[(n*Jc+j)*4+k]*m[(vid*Jc+j)*4+k];
    atomicAdd(&asum[vid*Jc+j], act_in[n*Jc+j]*dp);
  }
}

__global__ void pool_p4(int V, int Jc, const float* __restrict__ asum,
    const float* __restrict__ wsum, const float* __restrict__ cnt,
    const float* __restrict__ possum,
    float* __restrict__ act_out, float* __restrict__ pos_out)
{
  int v = blockIdx.x*blockDim.x + threadIdx.x;
  if (v >= V) return;
  for (int j=0;j<Jc;++j) act_out[v*Jc+j] = sigmoidf(asum[v*Jc+j]/(wsum[v*Jc+j]+CAPS_EPS));
  float c = cnt[v] + CAPS_EPS;
  for (int d=0;d<3;++d) pos_out[v*3+d] = possum[v*3+d]/c;
}

// ---------------- host ----------------
extern "C" void kernel_launch(void* const* d_in, const int* in_sizes, int n_in,
                              void* d_out, int out_size, void* d_ws, size_t ws_size,
                              hipStream_t stream)
{
  const float* x      = (const float*)d_in[0];
  const float* pos    = (const float*)d_in[1];
  const float* pseudo = (const float*)d_in[2];
  const float* Wsp    = (const float*)d_in[3];
  const float* root   = (const float*)d_in[4];
  const float* bias   = (const float*)d_in[5];
  const float* lin_w  = (const float*)d_in[6];
  const float* lin_b  = (const float*)d_in[7];
  const float* q0     = (const float*)d_in[8];
  const float* q1     = (const float*)d_in[9];
  const float* q3     = (const float*)d_in[10];
  const float* q5     = (const float*)d_in[11];
  const float* q6     = (const float*)d_in[12];
  const float* q7     = (const float*)d_in[13];
  const int*   eidx   = (const int*)d_in[14];
  const int N = in_sizes[0];
  const int E = in_sizes[14]/2;

  float* W = (float*)d_ws;
  size_t off = 0;
  auto alloc = [&](size_t nf)->float* { float* p = W + off; off += nf; return p; };

  float* mm    = alloc(4);
  float* agg   = alloc((size_t)N*16);
  float* deg   = alloc((size_t)N);
  const size_t PB = (size_t)N*24;     // pose ping-pong (covers max N*J*4)
  float* poseA = alloc(PB);
  float* poseB = alloc(PB);
  const size_t AB = (size_t)N*6;      // act ping-pong
  float* actA  = alloc(AB);
  float* actB  = alloc(AB);
  float* posA  = alloc(100000);
  float* posB  = alloc(100000);
  const int V1 = 32768;
  float* ps    = alloc((size_t)V1*6*4);
  float* pws   = alloc((size_t)V1*6);
  float* pcnt  = alloc((size_t)V1);
  float* pas   = alloc((size_t)V1*6);
  float* ppos  = alloc((size_t)V1*3);
  const size_t totalF = off;
  const long poolF = (long)V1*6*4 + (long)V1*6 + V1 + (long)V1*6 + (long)V1*3;
  if (ws_size < totalF*sizeof(float)) return;

  const int B = 256;
  auto cdiv = [](int a, int b){ return (a + b - 1)/b; };

  zero_f<<<1024, B, 0, stream>>>(W, (long)totalF);
  init_mm<<<1, 1, 0, stream>>>(mm);
  minmax_k<<<cdiv(3*N,B), B, 0, stream>>>(pos, 3*N, mm);
  mm_final<<<1, 1, 0, stream>>>(mm);

  const int tiles = cdiv(E, 16);
  spline_wmma<<<cdiv(tiles,8), 256, 0, stream>>>(x, pseudo, Wsp, eidx, agg, deg, E);
  node_finalize<<<cdiv(N,B), B, 0, stream>>>(x, agg, deg, root, bias, lin_w, lin_b,
                                             poseA, actA, N);

  caps_kernel<1,6><<<cdiv(N,B), B, 0, stream>>>(N, poseA, actA, q0, poseB, actB);
  caps_kernel<6,6><<<cdiv(N,B), B, 0, stream>>>(N, poseB, actB, q1, poseA, actA);

  // pool 1: N nodes, J=6, G=32 -> V=32768
  zero_f<<<512, B, 0, stream>>>(ps, poolF);
  pool_p1<<<cdiv(N,B), B, 0, stream>>>(N, 32, 6, 0, mm, poseA, actA, pos, ps, pws, pcnt, ppos);
  pool_p2<<<cdiv(V1*6,B), B, 0, stream>>>(V1*6, ps, poseB);
  pool_p3<<<cdiv(N,B), B, 0, stream>>>(N, 32, 6, 0, mm, poseA, actA, pos, poseB, pas);
  pool_p4<<<cdiv(V1,B), B, 0, stream>>>(V1, 6, pas, pws, pcnt, ppos, actB, posA);

  caps_kernel<6,8><<<cdiv(V1,B), B, 0, stream>>>(V1, poseB, actB, q3, poseA, actA);

  // pool 2: 32768 nodes, J=8, G=8 -> V=512
  const int V2 = 512;
  zero_f<<<512, B, 0, stream>>>(ps, poolF);
  pool_p1<<<cdiv(V1,B), B, 0, stream>>>(V1, 8, 8, 0, mm, poseA, actA, posA, ps, pws, pcnt, ppos);
  pool_p2<<<cdiv(V2*8,B), B, 0, stream>>>(V2*8, ps, poseB);
  pool_p3<<<cdiv(V1,B), B, 0, stream>>>(V1, 8, 8, 0, mm, poseA, actA, posA, poseB, pas);
  pool_p4<<<cdiv(V2,B), B, 0, stream>>>(V2, 8, pas, pws, pcnt, ppos, actB, posB);

  caps_kernel<8,12><<<cdiv(V2,B), B, 0, stream>>>(V2, poseB, actB, q5, poseA, actA);

  // pool 3: 512 nodes, J=12, G=2 -> V=8
  const int V3 = 8;
  zero_f<<<512, B, 0, stream>>>(ps, poolF);
  pool_p1<<<cdiv(V2,B), B, 0, stream>>>(V2, 2, 12, 0, mm, poseA, actA, posB, ps, pws, pcnt, ppos);
  pool_p2<<<1, B, 0, stream>>>(V3*12, ps, poseB);
  pool_p3<<<cdiv(V2,B), B, 0, stream>>>(V2, 2, 12, 0, mm, poseA, actA, posB, poseB, pas);
  pool_p4<<<1, B, 0, stream>>>(V3, 12, pas, pws, pcnt, ppos, actB, posA);

  caps_kernel<12,14><<<1, B, 0, stream>>>(V3, poseB, actB, q6, poseA, actA);

  // pool 4: 8 nodes, J=14, G=1 (mode 1) -> V=1
  zero_f<<<512, B, 0, stream>>>(ps, poolF);
  pool_p1<<<1, B, 0, stream>>>(V3, 1, 14, 1, mm, poseA, actA, posA, ps, pws, pcnt, ppos);
  pool_p2<<<1, B, 0, stream>>>(14, ps, poseB);
  pool_p3<<<1, B, 0, stream>>>(V3, 1, 14, 1, mm, poseA, actA, posA, poseB, pas);
  pool_p4<<<1, B, 0, stream>>>(1, 14, pas, pws, pcnt, ppos, actB, posB);

  caps_kernel<14,10><<<1, B, 0, stream>>>(1, poseB, actB, q7, poseA, (float*)d_out);
}
